// quantum_attention_6880537608586
// MI455X (gfx1250) — matrix operand order repaired and checked by reading, not compiled, not run on hardware
//
#include <hip/hip_runtime.h>
#include <hip/hip_bf16.h>

// ---- problem constants ----
#define B_   4
#define L_   2048
#define DM   1024
#define DK   128
#define MTOT (B_ * L_)          // 8192

typedef __attribute__((ext_vector_type(16))) _Float16 v16h;
typedef __attribute__((ext_vector_type(8)))  _Float16 v8h;
typedef __attribute__((ext_vector_type(8)))  float    v8f;
typedef __attribute__((ext_vector_type(4)))  float    v4f;

static __device__ inline v16h cat8(v8h lo, v8h hi) {
    v16h r;
#pragma unroll
    for (int i = 0; i < 8; ++i) { r[i] = lo[i]; r[i + 8] = hi[i]; }
    return r;
}

static __device__ inline v8f wmma_f16(v16h a, v16h b, v8f c) {
    // D = A(16x32 f16) * B(32x16 f16) + C(16x16 f32)
    return __builtin_amdgcn_wmma_f32_16x16x32_f16(
        /*neg_a=*/false, a, /*neg_b=*/false, b,
        /*c_mod=*/(short)0, c, /*reuse_a=*/false, /*reuse_b=*/false);
}

// ---------------------------------------------------------------------------
// Kernel 0: convert + transpose weights.  W[mat] is [DM][DK] f32 (row k, col n).
// WhT is [3][DK][DM] f16 so a WMMA B-fragment (fixed n, two 8-k runs) is two
// contiguous 16-byte runs.
// ---------------------------------------------------------------------------
__global__ __launch_bounds__(256) void wconv_kernel(
    const float* __restrict__ Wq, const float* __restrict__ Wk,
    const float* __restrict__ Wv, _Float16* __restrict__ WhT)
{
    int idx = blockIdx.x * 256 + threadIdx.x;       // 3*128*1024 total
    if (idx >= 3 * DK * DM) return;
    int mat = idx / (DK * DM);
    int rem = idx - mat * (DK * DM);
    int n   = rem / DM;
    int k   = rem - n * DM;
    const float* W = (mat == 0) ? Wq : (mat == 1) ? Wk : Wv;
    WhT[idx] = (_Float16)W[(size_t)k * DK + n];
}

// ---------------------------------------------------------------------------
// Kernel 1: fused Q/K/V projection.
//   grid = 512 blocks (16-row tiles of the 8192 rows), 256 threads = 8 waves.
//   Wave w computes output cols [16w, 16w+16) for all three matrices,
//   streaming k in 32-wide steps: 3 WMMAs per step, x converted f32->f16
//   in-register.  Q,K written row-major f16; V written transposed [b][dk][L].
// ---------------------------------------------------------------------------
__global__ __launch_bounds__(256) void proj_kernel(
    const float* __restrict__ x,
    const float* __restrict__ bq, const float* __restrict__ bk,
    const float* __restrict__ bv,
    const _Float16* __restrict__ WhT,     // [3][DK][DM]
    _Float16* __restrict__ Qh,            // [MTOT][DK]
    _Float16* __restrict__ Kh,            // [MTOT][DK]
    _Float16* __restrict__ VhT)           // [B_][DK][L_]
{
    const int lane = threadIdx.x & 31;
    const int wave = threadIdx.x >> 5;
    const int lm   = lane & 15;           // A: row m ; B: col n ; C: col n
    const int hi   = lane >> 4;           // lane-group select
    const int m0   = blockIdx.x * 16;     // global row tile
    const int n0   = wave * 16;           // dk col tile

    v8f acc[3];
#pragma unroll
    for (int t = 0; t < 3; ++t) acc[t] = (v8f){};

    const float* xrow = x + (size_t)(m0 + lm) * DM;

    for (int kt = 0; kt < DM; kt += 32) {
        // --- A fragment (16x32 f16): lane lm = row; two 8-half runs:
        //     k = kt + 8*hi + [0..7]  and  k = kt + 16 + 8*hi + [0..7]
        v4f f0 = *(const v4f*)(xrow + kt + hi * 8);
        v4f f1 = *(const v4f*)(xrow + kt + hi * 8 + 4);
        v4f f2 = *(const v4f*)(xrow + kt + 16 + hi * 8);
        v4f f3 = *(const v4f*)(xrow + kt + 16 + hi * 8 + 4);
        v16h a;
#pragma unroll
        for (int i = 0; i < 4; ++i) {
            a[i]      = (_Float16)f0[i];
            a[4 + i]  = (_Float16)f1[i];
            a[8 + i]  = (_Float16)f2[i];
            a[12 + i] = (_Float16)f3[i];
        }
        // --- B fragments (32x16): lane lm = col n; two 8-half runs at
        //     kt + 8*hi and kt + 16 + 8*hi, contiguous in WhT.
#pragma unroll
        for (int mat = 0; mat < 3; ++mat) {
            const _Float16* wp =
                WhT + ((size_t)(mat * DK + n0 + lm) * DM + kt + hi * 8);
            v16h b = cat8(*(const v8h*)wp, *(const v8h*)(wp + 16));
            acc[mat] = wmma_f16(a, b, acc[mat]);
        }
    }

    const float bias0 = bq[n0 + lm];
    const float bias1 = bk[n0 + lm];
    const float bias2 = bv[n0 + lm];

    const int b  = m0 / L_;
    const int ll = m0 % L_;

    // C layout: VGPR r holds (m = r + 8*hi, n = lm)
#pragma unroll
    for (int r = 0; r < 8; ++r) {
        int row = m0 + 8 * hi + r;
        Qh[(size_t)row * DK + n0 + lm] = (_Float16)(acc[0][r] + bias0);
        Kh[(size_t)row * DK + n0 + lm] = (_Float16)(acc[1][r] + bias1);
    }
    // V transposed: column (dk) fixed per lane, 8 consecutive l -> one b128.
    v8h vv;
#pragma unroll
    for (int r = 0; r < 8; ++r) vv[r] = (_Float16)(acc[2][r] + bias2);
    *(v8h*)(VhT + ((size_t)(b * DK + n0 + lm)) * L_ + ll + 8 * hi) = vv;
}

// ---------------------------------------------------------------------------
// Kernel 2: fused  out = cos(Q K^T) V  (flash-style, never materializes QK).
//   grid = 64 blocks (4 batches x 16), 256 threads = 8 waves.
//   Each wave owns 16 Q rows: preloads 4 Q A-fragments (k=0..127), then for
//   each 32-wide K/V tile: 8 WMMAs -> S(16x32) f32, native cos (v_cos_f32,
//   TRANS pipe, co-executes with the XDL WMMA pipe), f16, C->A layout shuffle
//   through a private LDS slab, 8 WMMAs accumulating O(16x128) f32.
//
//   Native-cos validity: |QK| ~ 40 rad here; v_cos_f32 is valid for
//   |arg|/2pi < 256 i.e. |QK| < ~1600, and its ~1e-5 error is negligible vs
//   the ~1e-2 rad phase error already implied by f16 Q/K storage.
// ---------------------------------------------------------------------------
__global__ __launch_bounds__(256) void cosattn_kernel(
    const _Float16* __restrict__ Qh, const _Float16* __restrict__ Kh,
    const _Float16* __restrict__ VhT, float* __restrict__ out)
{
    __shared__ __align__(16) _Float16 smem[8][16][32];   // 8 KB, 1 KB / wave

    const int lane = threadIdx.x & 31;
    const int wave = threadIdx.x >> 5;
    const int lm   = lane & 15;
    const int hi   = lane >> 4;
    const int b    = blockIdx.x >> 4;                    // 16 blocks per batch
    const int l0   = ((blockIdx.x & 15) * 8 + wave) * 16; // Q-row tile in batch

    // Preload Q A-fragments: aq[kt] covers k = [32*kt, 32*kt+32)
    v16h aq[4];
    const size_t qbase = ((size_t)b * L_ + l0 + lm) * DK;
#pragma unroll
    for (int kt = 0; kt < 4; ++kt) {
        const _Float16* qp = Qh + qbase + kt * 32 + hi * 8;
        aq[kt] = cat8(*(const v8h*)qp, *(const v8h*)(qp + 16));
    }

    v8f o[8];
#pragma unroll
    for (int nt = 0; nt < 8; ++nt) o[nt] = (v8f){};

    for (int lt = 0; lt < L_; lt += 32) {
        // ---- S = Q * K^T for K rows [lt, lt+32), two 16-col tiles ----
        v8f s[2];
        s[0] = (v8f){}; s[1] = (v8f){};
#pragma unroll
        for (int st = 0; st < 2; ++st) {
#pragma unroll
            for (int kt = 0; kt < 4; ++kt) {
                // B fragment: col n = K-row (lt + 16*st + lm), two 8-dk runs
                // at kt*32 + 8*hi and kt*32 + 16 + 8*hi in row-major Kh.
                const _Float16* kp =
                    Kh + ((size_t)b * L_ + lt + st * 16 + lm) * DK
                       + kt * 32 + hi * 8;
                v16h bk = cat8(*(const v8h*)kp, *(const v8h*)(kp + 16));
                s[st] = wmma_f16(aq[kt], bk, s[st]);
            }
        }

        // ---- cos(S) via native v_cos_f32 -> f16, C-layout -> LDS ----
#pragma unroll
        for (int st = 0; st < 2; ++st) {
#pragma unroll
            for (int r = 0; r < 8; ++r) {
                smem[wave][r + 8 * hi][st * 16 + lm] =
                    (_Float16)__cosf(s[st][r]);
            }
        }
        // Same-wave DS ops are in-order; dependence is through smem, so the
        // compiler inserts the s_wait_dscnt before the reads below.
        v16h sa;
        {
            const _Float16* sp0 = &smem[wave][lm][hi * 8];
            const _Float16* sp1 = &smem[wave][lm][16 + hi * 8];
            sa = cat8(*(const v8h*)sp0, *(const v8h*)sp1);
        }

        if (lt + 32 < L_) {  // uniform branch: prefetch next K/V tiles
            __builtin_prefetch(Kh + ((size_t)b * L_ + lt + 32 + lm) * DK, 0, 1);
            __builtin_prefetch(VhT + ((size_t)b * DK + lm) * L_ + lt + 32, 0, 1);
        }

        // ---- O += cos(S) * V : 8 n-tiles of dk ----
#pragma unroll
        for (int nt = 0; nt < 8; ++nt) {
            // B fragment: col n = dk (nt*16 + lm), two 8-l runs at
            // lt + 8*hi and lt + 16 + 8*hi in VhT.
            const _Float16* vp =
                VhT + ((size_t)(b * DK + nt * 16 + lm)) * L_ + lt + hi * 8;
            v16h bv = cat8(*(const v8h*)vp, *(const v8h*)(vp + 16));
            o[nt] = wmma_f16(sa, bv, o[nt]);
        }
    }

    // ---- epilogue: f32 store, row-major [b][l][dk] ----
#pragma unroll
    for (int nt = 0; nt < 8; ++nt) {
#pragma unroll
        for (int r = 0; r < 8; ++r) {
            out[((size_t)b * L_ + l0 + 8 * hi + r) * DK + nt * 16 + lm] =
                o[nt][r];
        }
    }
}

// ---------------------------------------------------------------------------
extern "C" void kernel_launch(void* const* d_in, const int* in_sizes, int n_in,
                              void* d_out, int out_size, void* d_ws,
                              size_t ws_size, hipStream_t stream)
{
    const float* x  = (const float*)d_in[0];
    const float* Wq = (const float*)d_in[1];
    const float* bq = (const float*)d_in[2];
    const float* Wk = (const float*)d_in[3];
    const float* bk = (const float*)d_in[4];
    const float* Wv = (const float*)d_in[5];
    const float* bv = (const float*)d_in[6];
    float* out = (float*)d_out;

    char* ws = (char*)d_ws;
    _Float16* Qh  = (_Float16*)(ws + (size_t)0);                 // 2 MB
    _Float16* Kh  = (_Float16*)(ws + (size_t)2 * 1024 * 1024);   // 2 MB
    _Float16* VhT = (_Float16*)(ws + (size_t)4 * 1024 * 1024);   // 2 MB
    _Float16* WhT = (_Float16*)(ws + (size_t)6 * 1024 * 1024);   // 0.75 MB

    // 1) weights -> f16, transposed for contiguous B-fragments
    wconv_kernel<<<(3 * DK * DM + 255) / 256, 256, 0, stream>>>(Wq, Wk, Wv, WhT);
    // 2) fused Q/K/V projection (WMMA f16, f32 accumulate)
    proj_kernel<<<MTOT / 16, 256, 0, stream>>>(x, bq, bk, bv, WhT, Qh, Kh, VhT);
    // 3) fused cos-attention: out = cos(Q K^T) V
    cosattn_kernel<<<B_ * (L_ / 128), 256, 0, stream>>>(Qh, Kh, VhT, out);
}
